// GraphSAGEEncoder_5755256176689
// MI455X (gfx1250) — compile-verified, matrix operand order checked by reading, not run on hardware
//
#include <hip/hip_runtime.h>
#include <cstdint>
#include <cstddef>

// ---------------------------------------------------------------------------
// GraphSAGE encoder (3 layers, mean aggregation) + L2 norm + linear head.
// Memory-bound on MI455X; GEMMs run on the WMMA pipe (f16 in, f32 acc).
// K/M/D are compile-time to get b64/b128 loads and constant-folded addressing.
// ---------------------------------------------------------------------------

typedef __attribute__((ext_vector_type(16))) _Float16 v16h;
typedef __attribute__((ext_vector_type(8)))  float    v8f;

union AFrag { v16h v; _Float16 e[16]; };
union CFrag { v8f  v; float    e[8];  };

#define N_NODES 100000
#define N_EDGES 600000

// ----------------------------- utility kernels -----------------------------

__global__ void zero_f32(float* __restrict__ p, int n) {
  int i = blockIdx.x * blockDim.x + threadIdx.x;
  if (i < n) p[i] = 0.0f;
}

__global__ void degree_kernel(const long long* __restrict__ ei,
                              float* __restrict__ deg, int E) {
  int e = blockIdx.x * blockDim.x + threadIdx.x;
  if (e < E) {
    int d = (int)ei[E + e];           // dst row of edge_index
    atomicAdd(&deg[d], 1.0f);
  }
}

__global__ void rdeg_kernel(float* __restrict__ deg, int n) {
  int i = blockIdx.x * blockDim.x + threadIdx.x;
  if (i < n) deg[i] = 1.0f / fmaxf(deg[i], 1.0f);   // mean = agg * rdeg
}

// scatter-add h[src,f] into agg[dst,f]; one thread per (edge, feature).
// D is compile-time: idx/D becomes a mul-shift, addressing folds to shifts.
template <int D>
__global__ void edge_agg_kernel(const long long* __restrict__ ei,
                                const float* __restrict__ h,
                                float* __restrict__ agg, int E) {
  int idx = blockIdx.x * blockDim.x + threadIdx.x;   // < E*D (max 76.8M)
  if (idx >= E * D) return;
  int e = idx / D;
  int f = idx - e * D;
  long long s = ei[e];
  long long d = ei[E + e];
  atomicAdd(&agg[(size_t)d * D + f], h[(size_t)s * D + f]);
}

// ------------------------- fused SAGE layer (WMMA) -------------------------
// out[N,M] = BN( mean @ Wl + b + hself @ Wr ) [optional ReLU]
// One wave computes one 16x16 output tile. K padded to 32 with zeros.
template <int K, int M, int RELU>
__global__ __launch_bounds__(32) void sage_gemm_wmma(
    const float* __restrict__ agg,    // [N,K] neighbor sums
    const float* __restrict__ rdeg,   // [N]   1/max(deg,1)
    const float* __restrict__ hself,  // [N,K] self features
    const float* __restrict__ Wl,     // [K,M]
    const float* __restrict__ Wr,     // [K,M]
    const float* __restrict__ bias,   // [M]
    const float* __restrict__ bn_g, const float* __restrict__ bn_b,
    const float* __restrict__ bn_m, const float* __restrict__ bn_v,
    float* __restrict__ out) {        // [N,M]
  const int lane = threadIdx.x;       // wave32
  const int half = lane >> 4;         // 0: K-sub lo, 1: K-sub hi
  const int idx16 = lane & 15;        // A: row-in-tile, B: col-in-tile
  const int rt = blockIdx.x;          // row tile (N/16 = 6250 exact)
  const int ct = blockIdx.y;          // col tile
  const int row = rt * 16 + idx16;    // A fragment row for this lane
  const int col = ct * 16 + idx16;    // B fragment col for this lane

  const float  rd   = rdeg[row];
  const float* aRow = agg + (size_t)row * K;
  const float* hRow = hself + (size_t)row * K;
  const float* wl   = Wl + col;       // column `col`, stride M
  const float* wr   = Wr + col;

  CFrag c;
#pragma unroll
  for (int r = 0; r < 8; ++r) c.e[r] = 0.0f;

  constexpr int KB = (K + 31) / 32;
#pragma unroll
  for (int kb = 0; kb < KB; ++kb) {
    AFrag am, ah, bl, br;
#pragma unroll
    for (int i = 0; i < 8; ++i) {
      // ISA 16-bit A-layout: lanes 0-15 VGPR i -> K = 2i (i<4) / 16+2(i-4),
      // lanes 16-31 shifted by +8 within each half. kk is always even.
      const int kk = kb * 32 +
                     ((i < 4) ? (8 * half + 2 * i) : (16 + 8 * half + 2 * (i - 4)));
      float a0 = 0.f, a1 = 0.f, h0 = 0.f, h1 = 0.f;
      float w0 = 0.f, w1 = 0.f, u0 = 0.f, u1 = 0.f;
      if constexpr (K % 32 == 0) {
        // always a full, 8B-aligned pair -> global_load_b64
        const float2 av = *(const float2*)(aRow + kk);
        const float2 hv = *(const float2*)(hRow + kk);
        a0 = av.x * rd; a1 = av.y * rd;
        h0 = hv.x;      h1 = hv.y;
        w0 = wl[(size_t)kk * M];       w1 = wl[(size_t)(kk + 1) * M];
        u0 = wr[(size_t)kk * M];       u1 = wr[(size_t)(kk + 1) * M];
      } else {
        if (kk + 1 < K) {              // full pair (kk even => 8B aligned)
          const float2 av = *(const float2*)(aRow + kk);
          const float2 hv = *(const float2*)(hRow + kk);
          a0 = av.x * rd; a1 = av.y * rd;
          h0 = hv.x;      h1 = hv.y;
          w0 = wl[(size_t)kk * M];     w1 = wl[(size_t)(kk + 1) * M];
          u0 = wr[(size_t)kk * M];     u1 = wr[(size_t)(kk + 1) * M];
        } else if (kk < K) {           // lone trailing element
          a0 = aRow[kk] * rd;
          h0 = hRow[kk];
          w0 = wl[(size_t)kk * M];
          u0 = wr[(size_t)kk * M];
        }
      }
      am.e[2 * i] = (_Float16)a0; am.e[2 * i + 1] = (_Float16)a1;
      ah.e[2 * i] = (_Float16)h0; ah.e[2 * i + 1] = (_Float16)h1;
      bl.e[2 * i] = (_Float16)w0; bl.e[2 * i + 1] = (_Float16)w1;
      br.e[2 * i] = (_Float16)u0; br.e[2 * i + 1] = (_Float16)u1;
    }
    // mean @ Wl  and  hself @ Wr  accumulate into the same f32 tile
    c.v = __builtin_amdgcn_wmma_f32_16x16x32_f16(
        false, am.v, false, bl.v, (short)0, c.v, false, false);
    c.v = __builtin_amdgcn_wmma_f32_16x16x32_f16(
        false, ah.v, false, br.v, (short)0, c.v, false, false);
  }

  // epilogue: +bias, BN (eval), optional ReLU.  C/D layout:
  // lanes 0-15: VGPR r -> row r; lanes 16-31: VGPR r -> row r+8; col = lane&15
  const float bs    = bias[col];
  const float scale = bn_g[col] * rsqrtf(bn_v[col] + 1e-5f);
  const float mm    = bn_m[col];
  const float bb    = bn_b[col];
#pragma unroll
  for (int r = 0; r < 8; ++r) {
    const int orow = rt * 16 + r + 8 * half;
    float y = c.e[r] + bs;
    y = (y - mm) * scale + bb;
    if (RELU) y = fmaxf(y, 0.0f);
    out[(size_t)orow * M + col] = y;
  }
}

// -------------------- L2 normalize + 2-class linear head -------------------
// one wave per node; lane handles 4 of 128 features
__global__ __launch_bounds__(256) void norm_logits_kernel(
    const float* __restrict__ h3,     // [N,128]
    const float* __restrict__ Wc,     // [128,2]
    const float* __restrict__ bc,     // [2]
    float* __restrict__ out, int n) { // emb [N,128] then logits [N,2]
  const int lane = threadIdx.x & 31;
  const int node = blockIdx.x * 8 + (threadIdx.x >> 5);
  if (node >= n) return;

  const float4 v = ((const float4*)(h3 + (size_t)node * 128))[lane];
  float s = v.x * v.x + v.y * v.y + v.z * v.z + v.w * v.w;
#pragma unroll
  for (int off = 16; off > 0; off >>= 1) s += __shfl_xor(s, off, 32);
  const float scale = 1.0f / fmaxf(sqrtf(s), 1e-12f);

  float4 e4;
  e4.x = v.x * scale; e4.y = v.y * scale; e4.z = v.z * scale; e4.w = v.w * scale;
  ((float4*)(out + (size_t)node * 128))[lane] = e4;

  const int f0 = lane * 4;
  float p0 = e4.x * Wc[(f0 + 0) * 2 + 0] + e4.y * Wc[(f0 + 1) * 2 + 0] +
             e4.z * Wc[(f0 + 2) * 2 + 0] + e4.w * Wc[(f0 + 3) * 2 + 0];
  float p1 = e4.x * Wc[(f0 + 0) * 2 + 1] + e4.y * Wc[(f0 + 1) * 2 + 1] +
             e4.z * Wc[(f0 + 2) * 2 + 1] + e4.w * Wc[(f0 + 3) * 2 + 1];
#pragma unroll
  for (int off = 16; off > 0; off >>= 1) {
    p0 += __shfl_xor(p0, off, 32);
    p1 += __shfl_xor(p1, off, 32);
  }
  if (lane == 0) {
    float* lg = out + (size_t)n * 128 + (size_t)node * 2;
    lg[0] = p0 + bc[0];
    lg[1] = p1 + bc[1];
  }
}

// --------------------------------- driver ----------------------------------

extern "C" void kernel_launch(void* const* d_in, const int* in_sizes, int n_in,
                              void* d_out, int out_size, void* d_ws,
                              size_t ws_size, hipStream_t stream) {
  const float*     x  = (const float*)d_in[0];
  const long long* ei = (const long long*)d_in[1];    // int64 [2, E]
  const float* W1l = (const float*)d_in[2];
  const float* W1r = (const float*)d_in[3];
  const float* b1  = (const float*)d_in[4];
  const float* g1  = (const float*)d_in[5];
  const float* bb1 = (const float*)d_in[6];
  const float* m1  = (const float*)d_in[7];
  const float* v1  = (const float*)d_in[8];
  const float* W2l = (const float*)d_in[9];
  const float* W2r = (const float*)d_in[10];
  const float* b2  = (const float*)d_in[11];
  const float* g2  = (const float*)d_in[12];
  const float* bb2 = (const float*)d_in[13];
  const float* m2  = (const float*)d_in[14];
  const float* v2  = (const float*)d_in[15];
  const float* W3l = (const float*)d_in[16];
  const float* W3r = (const float*)d_in[17];
  const float* b3  = (const float*)d_in[18];
  const float* g3  = (const float*)d_in[19];
  const float* bb3 = (const float*)d_in[20];
  const float* m3  = (const float*)d_in[21];
  const float* v3  = (const float*)d_in[22];
  const float* Wc  = (const float*)d_in[23];
  const float* bc  = (const float*)d_in[24];

  const int N = N_NODES, E = N_EDGES;

  // workspace carve-out (256B-aligned slabs)
  char* ws = (char*)d_ws;
  size_t off = 0;
  auto carve = [&](size_t nf) {
    float* p = (float*)(ws + off);
    off += ((nf * sizeof(float) + 255) / 256) * 256;
    return p;
  };
  float* deg = carve((size_t)N);            // degree -> reciprocal, in place
  float* agg = carve((size_t)N * 128);      // neighbor-sum buffer (reused)
  float* h1  = carve((size_t)N * 64);
  float* h2  = carve((size_t)N * 128);
  float* h3  = carve((size_t)N * 128);
  (void)ws_size; (void)in_sizes; (void)n_in; (void)out_size;

  const dim3 B(256);
  auto blocks = [](int n) { return dim3((unsigned)((n + 255) / 256)); };

  // degree (graph is static across layers) -> 1/max(deg,1)
  zero_f32<<<blocks(N), B, 0, stream>>>(deg, N);
  degree_kernel<<<blocks(E), B, 0, stream>>>(ei, deg, E);
  rdeg_kernel<<<blocks(N), B, 0, stream>>>(deg, N);

  // ---- layer 1: [N,10] -> [N,64], BN+ReLU ----
  zero_f32<<<blocks(N * 10), B, 0, stream>>>(agg, N * 10);
  edge_agg_kernel<10><<<blocks(E * 10), B, 0, stream>>>(ei, x, agg, E);
  sage_gemm_wmma<10, 64, 1><<<dim3(N / 16, 64 / 16), dim3(32), 0, stream>>>(
      agg, deg, x, W1l, W1r, b1, g1, bb1, m1, v1, h1);

  // ---- layer 2: [N,64] -> [N,128], BN+ReLU ----
  zero_f32<<<blocks(N * 64), B, 0, stream>>>(agg, N * 64);
  edge_agg_kernel<64><<<blocks(E * 64), B, 0, stream>>>(ei, h1, agg, E);
  sage_gemm_wmma<64, 128, 1><<<dim3(N / 16, 128 / 16), dim3(32), 0, stream>>>(
      agg, deg, h1, W2l, W2r, b2, g2, bb2, m2, v2, h2);

  // ---- layer 3: [N,128] -> [N,128], BN (no ReLU) ----
  zero_f32<<<blocks(N * 128), B, 0, stream>>>(agg, N * 128);
  edge_agg_kernel<128><<<blocks(E * 128), B, 0, stream>>>(ei, h2, agg, E);
  sage_gemm_wmma<128, 128, 0><<<dim3(N / 16, 128 / 16), dim3(32), 0, stream>>>(
      agg, deg, h2, W3l, W3r, b3, g3, bb3, m3, v3, h3);

  // ---- L2 normalize + classifier head -> d_out ----
  norm_logits_kernel<<<dim3((N + 7) / 8), dim3(256), 0, stream>>>(
      h3, Wc, bc, (float*)d_out, N);
}
